// LabelAwareContrastiveLoss_86784109183051
// MI455X (gfx1250) — compile-verified, matrix operand order checked by reading, not run on hardware
//
#include <hip/hip_runtime.h>
#include <math.h>

// ---------------------------------------------------------------------------
// LabelAwareContrastiveLoss for MI455X (gfx1250, wave32, WMMA).
//   B = 4096, D = 256, labels in {0,1}, out = scalar f32.
// Workspace layout (floats): sim[4096*4096] | sq[8192] | posL[4096]
//                            | hardN[4096] | tripPer[8192] | cnt[2] (ints)
// Requires ws_size >= ~68 MB.
// ---------------------------------------------------------------------------

#define BDIM 4096
#define DDIM 256
#define NROW 8192           // 2B
#define HARD_K 819          // int(4096*0.2)
#define TEMP 0.07f
#define MARGIN 0.5f
#define HALFCNT 33554432u   // (2B)^2 / 2 = 8192*8192/2

typedef __attribute__((ext_vector_type(2)))  float v2f;
typedef __attribute__((ext_vector_type(8)))  float v8f;
typedef __attribute__((ext_vector_type(16))) __bf16 v16bf;

// ---------------- threefry-2x32 (JAX PRNG), constexpr so keys fold ---------
struct U2 { unsigned a, b; };

__host__ __device__ constexpr U2 threefry2x32(unsigned k0, unsigned k1,
                                              unsigned x0, unsigned x1) {
  const unsigned ks2 = k0 ^ k1 ^ 0x1BD11BDAu;
  unsigned ks[3] = {k0, k1, ks2};
  const unsigned r0[4] = {13u, 15u, 26u, 6u};
  const unsigned r1[4] = {17u, 29u, 16u, 24u};
  x0 += k0; x1 += k1;
  for (int gp = 0; gp < 5; ++gp) {
    for (int r = 0; r < 4; ++r) {
      unsigned rot = (gp & 1) ? r1[r] : r0[r];
      x0 += x1;
      x1 = (x1 << rot) | (x1 >> (32u - rot));
      x1 ^= x0;
    }
    x0 += ks[(gp + 1) % 3];
    x1 += ks[(gp + 2) % 3] + (unsigned)(gp + 1);
  }
  return U2{x0, x1};
}

// jax.random.key(42) -> [0,42]; split -> counts arange(4): blocks (0,2),(1,3)
// kp = (w0(0,2), w0(1,3)) ; kn = (w1(0,2), w1(1,3))
constexpr U2 TF_B0 = threefry2x32(0u, 42u, 0u, 2u);
constexpr U2 TF_B1 = threefry2x32(0u, 42u, 1u, 3u);
constexpr unsigned KP0 = TF_B0.a, KP1 = TF_B1.a;
constexpr unsigned KN0 = TF_B0.b, KN1 = TF_B1.b;

__device__ __forceinline__ float gumbel_from_bits(unsigned bits) {
  // jax.random.uniform(minval=tiny, maxval=1.0) then -log(-log(u))
  const float TINY = 1.17549435e-38f;
  float u01 = __uint_as_float((bits >> 9) | 0x3F800000u) - 1.0f;  // [0,1)
  float u = fmaxf(TINY, u01 * (1.0f - TINY) + TINY);
  return -logf(-logf(u));
}

__device__ __forceinline__ float softplusf(float x) {
  // jnp.logaddexp(x, 0) = max(x,0) + log1p(exp(-|x|))
  return fmaxf(x, 0.0f) + log1pf(expf(-fabsf(x)));
}

__device__ __forceinline__ unsigned f2key(float f) {   // order-preserving map
  unsigned b = __float_as_uint(f);
  return (b & 0x80000000u) ? ~b : (b | 0x80000000u);
}
__device__ __forceinline__ float key2f(unsigned k) {
  unsigned b = (k & 0x80000000u) ? (k & 0x7FFFFFFFu) : ~k;
  return __uint_as_float(b);
}

// ---------------- block reductions (256 threads, deterministic tree) -------
#define REDSUM_F(arr, var) do { (arr)[t] = (var); __syncthreads();            \
  for (int s_ = 128; s_ > 0; s_ >>= 1) {                                      \
    if (t < s_) (arr)[t] += (arr)[t + s_]; __syncthreads(); }                 \
  (var) = (arr)[0]; __syncthreads(); } while (0)

#define REDSUM_I(arr, var) do { (arr)[t] = (var); __syncthreads();            \
  for (int s_ = 128; s_ > 0; s_ >>= 1) {                                      \
    if (t < s_) (arr)[t] += (arr)[t + s_]; __syncthreads(); }                 \
  (var) = (arr)[0]; __syncthreads(); } while (0)

#define REDMAX_F(arr, var) do { (arr)[t] = (var); __syncthreads();            \
  for (int s_ = 128; s_ > 0; s_ >>= 1) {                                      \
    if (t < s_) (arr)[t] = fmaxf((arr)[t], (arr)[t + s_]); __syncthreads(); } \
  (var) = (arr)[0]; __syncthreads(); } while (0)

#define REDARGMAX(va, ia, v, ix) do { (va)[t] = (v); (ia)[t] = (ix);          \
  __syncthreads();                                                            \
  for (int s_ = 128; s_ > 0; s_ >>= 1) {                                      \
    if (t < s_) { float ov = (va)[t + s_]; int oi = (ia)[t + s_];             \
      if (ov > (va)[t] || (ov == (va)[t] && oi < (ia)[t])) {                  \
        (va)[t] = ov; (ia)[t] = oi; } }                                       \
    __syncthreads(); }                                                        \
  (v) = (va)[0]; (ix) = (ia)[0]; __syncthreads(); } while (0)

// ---------------- kernel 1: label counts -----------------------------------
__global__ __launch_bounds__(256) void count_kernel(const int* __restrict__ labels,
                                                    int* __restrict__ cnt) {
  __shared__ int red[256];
  const int t = threadIdx.x;
  int c0 = 0;
  for (int j = t; j < BDIM; j += 256) c0 += (labels[j] == 0);
  REDSUM_I(red, c0);
  if (t == 0) { cnt[0] = c0; cnt[1] = BDIM - c0; }
}

// ---------------- kernel 2: squared norms of all 8192 rows -----------------
__global__ __launch_bounds__(256) void sq_kernel(const float* __restrict__ mic,
                                                 const float* __restrict__ fm,
                                                 float* __restrict__ sq) {
  __shared__ float red[256];
  const int t = threadIdx.x;
  const int r = blockIdx.x;
  const float* p = (r < BDIM) ? (mic + (size_t)r * DDIM)
                              : (fm + (size_t)(r - BDIM) * DDIM);
  float v = p[t];
  float s = v * v;
  REDSUM_F(red, s);
  if (t == 0) sq[r] = s;
}

// ---------------- kernel 3: WMMA GEMM  sim = (mic @ fmᵀ)/0.07 --------------
// grid = (4096/16, 4096/128), block = 256 (8 waves). Wave w owns cols
// n0 = by*128 + 16*w. f32 16x16x4 fragments: lane L holds row (base + L%16),
// cols k0 + 2*(L/16) .. +1  -> one contiguous b64 load per fragment per step.
__global__ __launch_bounds__(256) void gemm_kernel(const float* __restrict__ mic,
                                                   const float* __restrict__ fm,
                                                   float* __restrict__ sim) {
  const int lane = threadIdx.x & 31;
  const int wv   = threadIdx.x >> 5;
  const int m0 = blockIdx.x * 16;
  const int n0 = blockIdx.y * 128 + wv * 16;
  const int g  = lane >> 4;          // 0 or 1 (half-wave)
  const int ln = lane & 15;

  const float* Arow = mic + (size_t)(m0 + ln) * DDIM + 2 * g;
  const float* Brow = fm  + (size_t)(n0 + ln) * DDIM + 2 * g;

#if __has_builtin(__builtin_amdgcn_wmma_f32_16x16x4_f32)
  v8f a0 = {}, a1 = {}, a2 = {}, a3 = {};
#pragma unroll 4
  for (int k = 0; k < DDIM; k += 16) {
    a0 = __builtin_amdgcn_wmma_f32_16x16x4_f32(false, *(const v2f*)(Arow + k),
            false, *(const v2f*)(Brow + k),      (short)0, a0, false, false);
    a1 = __builtin_amdgcn_wmma_f32_16x16x4_f32(false, *(const v2f*)(Arow + k + 4),
            false, *(const v2f*)(Brow + k + 4),  (short)0, a1, false, false);
    a2 = __builtin_amdgcn_wmma_f32_16x16x4_f32(false, *(const v2f*)(Arow + k + 8),
            false, *(const v2f*)(Brow + k + 8),  (short)0, a2, false, false);
    a3 = __builtin_amdgcn_wmma_f32_16x16x4_f32(false, *(const v2f*)(Arow + k + 12),
            false, *(const v2f*)(Brow + k + 12), (short)0, a3, false, false);
  }
  v8f acc = (a0 + a1) + (a2 + a3);
#else
  // bf16 fallback (codegen-confirmed builtin). 16-bit A 16x32 layout:
  // VGPR r holds K = (r<4 ? 2r : 16+2(r-4)) + 8g  and K+1.
  const float* ArowB = mic + (size_t)(m0 + ln) * DDIM;
  const float* BrowB = fm  + (size_t)(n0 + ln) * DDIM;
  v8f b0 = {}, b1 = {};
#pragma unroll 2
  for (int k = 0; k < DDIM; k += 64) {
    v16bf fa0, fb0, fa1, fb1;
#pragma unroll
    for (int r = 0; r < 8; ++r) {
      int K = ((r < 4) ? (2 * r) : (16 + 2 * (r - 4))) + 8 * g;
      fa0[2 * r]     = (__bf16)ArowB[k + K];
      fa0[2 * r + 1] = (__bf16)ArowB[k + K + 1];
      fb0[2 * r]     = (__bf16)BrowB[k + K];
      fb0[2 * r + 1] = (__bf16)BrowB[k + K + 1];
      fa1[2 * r]     = (__bf16)ArowB[k + 32 + K];
      fa1[2 * r + 1] = (__bf16)ArowB[k + 32 + K + 1];
      fb1[2 * r]     = (__bf16)BrowB[k + 32 + K];
      fb1[2 * r + 1] = (__bf16)BrowB[k + 32 + K + 1];
    }
    b0 = __builtin_amdgcn_wmma_f32_16x16x32_bf16(false, fa0, false, fb0,
                                                 (short)0, b0, false, false);
    b1 = __builtin_amdgcn_wmma_f32_16x16x32_bf16(false, fa1, false, fb1,
                                                 (short)0, b1, false, false);
  }
  v8f acc = b0 + b1;
#endif

  // C/D layout: vgpr v -> row m0 + v + 8g, col n0 + lane%16
#pragma unroll
  for (int v = 0; v < 8; ++v) {
    int r = m0 + v + 8 * g;
    sim[(size_t)r * BDIM + n0 + ln] = acc[v] / TEMP;
  }
}

// ---------------- kernel 4: per-row positive loss + exact top-k softplus ---
__global__ __launch_bounds__(256) void rowreduce_kernel(const float* __restrict__ sim,
                                                        const int* __restrict__ labels,
                                                        const int* __restrict__ cnt,
                                                        float* __restrict__ posL,
                                                        float* __restrict__ hardN) {
  __shared__ float row[BDIM];
  __shared__ unsigned ukey[BDIM];
  __shared__ char lab[BDIM];
  __shared__ float redf[256];
  __shared__ int   redi[256];
  const int t = threadIdx.x;
  const int i = blockIdx.x;

  for (int j = t; j < BDIM; j += 256) {
    row[j] = sim[(size_t)i * BDIM + j];
    lab[j] = (char)labels[j];
  }
  __syncthreads();
  const char myLab = lab[i];

  // masked values (neg_mask keeps sim, everything else exactly 0) -> sortkeys
  for (int j = t; j < BDIM; j += 256) {
    float v = (lab[j] != myLab) ? row[j] : 0.0f;
    ukey[j] = f2key(v);
  }
  __syncthreads();

  // --- logsumexp + positive sum ---
  float m = -INFINITY;
  for (int j = t; j < BDIM; j += 256) m = fmaxf(m, row[j]);
  REDMAX_F(redf, m);
  float se = 0.0f, ps = 0.0f;
  for (int j = t; j < BDIM; j += 256) {
    float v = row[j];
    se += expf(v - m);
    if (lab[j] == myLab && j != i) ps += v;
  }
  REDSUM_F(redf, se);
  REDSUM_F(redf, ps);
  const float lse = m + logf(se);
  const float cntpos = (float)((myLab ? cnt[1] : cnt[0]) - 1);
  if (t == 0) posL[i] = -(ps - cntpos * lse) / (cntpos + 1e-8f);

  // --- exact k-th largest via binary search on order-preserving uint key ---
  unsigned ans = 0u;
  for (int b = 31; b >= 0; --b) {
    const unsigned cand = ans | (1u << b);
    int c = 0;
    for (int j = t; j < BDIM; j += 256) c += (ukey[j] >= cand);
    REDSUM_I(redi, c);
    if (c >= HARD_K) ans = cand;          // uniform decision, c broadcast
  }
  // sum softplus over strictly-greater keys; fill remainder with threshold
  float ssum = 0.0f;
  int cgt = 0;
  for (int j = t; j < BDIM; j += 256) {
    unsigned k = ukey[j];
    if (k > ans) { ssum += softplusf(key2f(k)); ++cgt; }
  }
  REDSUM_F(redf, ssum);
  REDSUM_I(redi, cgt);
  if (t == 0) hardN[i] = ssum + (float)(HARD_K - cgt) * softplusf(key2f(ans));
}

// ---------------- kernel 5: triplet loss (JAX-exact Gumbel-max sampling) ---
// One block per row-pair (i, i+4096): one threefry block yields both rows'
// gumbels (word0 -> row i, word1 -> row i+4096), halving RNG work.
__global__ __launch_bounds__(256) void triplet_kernel(const float* __restrict__ mic,
                                                      const float* __restrict__ fm,
                                                      const int* __restrict__ labels,
                                                      const float* __restrict__ sq,
                                                      const int* __restrict__ cnt,
                                                      float* __restrict__ tripPer) {
  __shared__ char lab[BDIM];
  __shared__ float redf[256];
  __shared__ int   redi[256];
  const int t = threadIdx.x;
  const int i = blockIdx.x;          // row i and row i+4096
  for (int j = t; j < BDIM; j += 256) lab[j] = (char)labels[j];
  __syncthreads();
  const char myLab = lab[i];

  // best[0]=pos row i, best[1]=neg row i, best[2]=pos row i2, best[3]=neg row i2
  float bv[4] = {-INFINITY, -INFINITY, -INFINITY, -INFINITY};
  int   bi[4] = {0x7FFFFFFF, 0x7FFFFFFF, 0x7FFFFFFF, 0x7FFFFFFF};
  const unsigned base = (unsigned)i * (unsigned)NROW;

  for (int j = t; j < NROW; j += 256) {
    const unsigned lin = base + (unsigned)j;     // < HALFCNT always (i < 4096)
    const bool eq = (lab[j & (BDIM - 1)] == myLab);
    if (eq) {
      U2 gp = threefry2x32(KP0, KP1, lin, lin + HALFCNT);
      if (j != i) {                               // pos candidate for row i
        float g = gumbel_from_bits(gp.a);
        if (g > bv[0]) { bv[0] = g; bi[0] = j; }
      }
      if (j != i + BDIM) {                        // pos candidate for row i2
        float g = gumbel_from_bits(gp.b);
        if (g > bv[2]) { bv[2] = g; bi[2] = j; }
      }
    } else {
      U2 gn = threefry2x32(KN0, KN1, lin, lin + HALFCNT);
      float g0 = gumbel_from_bits(gn.a);
      if (g0 > bv[1]) { bv[1] = g0; bi[1] = j; }
      float g1 = gumbel_from_bits(gn.b);
      if (g1 > bv[3]) { bv[3] = g1; bi[3] = j; }
    }
  }
  REDARGMAX(redf, redi, bv[0], bi[0]);
  REDARGMAX(redf, redi, bv[1], bi[1]);
  REDARGMAX(redf, redi, bv[2], bi[2]);
  REDARGMAX(redf, redi, bv[3], bi[3]);
#pragma unroll
  for (int s = 0; s < 4; ++s) if (bi[s] >= NROW) bi[s] = 0;  // argmax(all -inf)=0

  // distances: d = sqrt(max(sq_a + sq_b - 2*dot, 1e-12)), matching reference
  const int rA[4] = {i, i, i + BDIM, i + BDIM};
  float dst[4];
#pragma unroll
  for (int s = 0; s < 4; ++s) {
    const int ra = rA[s], rb = bi[s];
    const float* pa = (ra < BDIM) ? (mic + (size_t)ra * DDIM)
                                  : (fm + (size_t)(ra - BDIM) * DDIM);
    const float* pb = (rb < BDIM) ? (mic + (size_t)rb * DDIM)
                                  : (fm + (size_t)(rb - BDIM) * DDIM);
    float p = pa[t] * pb[t];
    REDSUM_F(redf, p);
    float d2 = sq[ra] + sq[rb] - 2.0f * p;
    dst[s] = sqrtf(fmaxf(d2, 1e-12f));
  }
  const bool has = ((myLab ? cnt[0] : cnt[1]) > 0);  // opposite label exists
  if (t == 0) {
    tripPer[i]        = has ? fmaxf(dst[0] - dst[1] + MARGIN, 0.0f) : 0.0f;
    tripPer[i + BDIM] = has ? fmaxf(dst[2] - dst[3] + MARGIN, 0.0f) : 0.0f;
  }
}

// ---------------- kernel 6: final deterministic reduction ------------------
__global__ __launch_bounds__(256) void finalize_kernel(const float* __restrict__ posL,
                                                       const float* __restrict__ hardN,
                                                       const float* __restrict__ tripPer,
                                                       const int* __restrict__ cnt,
                                                       float* __restrict__ out) {
  __shared__ float red[256];
  const int t = threadIdx.x;
  float a = 0.0f, b = 0.0f, c = 0.0f;
  for (int j = t; j < BDIM; j += 256) a += posL[j];
  for (int j = t; j < BDIM; j += 256) b += hardN[j];
  for (int j = t; j < NROW; j += 256) c += tripPer[j];
  REDSUM_F(red, a);
  REDSUM_F(red, b);
  REDSUM_F(red, c);
  if (t == 0) {
    const float n = (cnt[0] > 0 && cnt[1] > 0) ? (float)NROW : 1.0f;
    out[0] = a / (float)BDIM + b / ((float)BDIM * (float)HARD_K) + c / n;
  }
}

// ---------------------------------------------------------------------------
extern "C" void kernel_launch(void* const* d_in, const int* in_sizes, int n_in,
                              void* d_out, int out_size, void* d_ws, size_t ws_size,
                              hipStream_t stream) {
  const float* mic    = (const float*)d_in[0];
  const float* fm     = (const float*)d_in[1];
  const int*   labels = (const int*)d_in[2];

  float* wsf   = (float*)d_ws;
  float* sim   = wsf;                              // 4096*4096
  float* sq    = sim + (size_t)BDIM * BDIM;        // 8192
  float* posL  = sq + NROW;                        // 4096
  float* hardN = posL + BDIM;                      // 4096
  float* tripP = hardN + BDIM;                     // 8192
  int*   cnt   = (int*)(tripP + NROW);             // 2

  count_kernel  <<<1, 256, 0, stream>>>(labels, cnt);
  sq_kernel     <<<NROW, 256, 0, stream>>>(mic, fm, sq);
  gemm_kernel   <<<dim3(BDIM / 16, BDIM / 128), 256, 0, stream>>>(mic, fm, sim);
  rowreduce_kernel<<<BDIM, 256, 0, stream>>>(sim, labels, cnt, posL, hardN);
  triplet_kernel<<<BDIM, 256, 0, stream>>>(mic, fm, labels, sq, cnt, tripP);
  finalize_kernel<<<1, 256, 0, stream>>>(posL, hardN, tripP, cnt, (float*)d_out);
}
